// STULayer_85392539779902
// MI455X (gfx1250) — matrix operand order repaired and checked by reading, not compiled, not run on hardware
//
#include <hip/hip_runtime.h>
#include <hip/hip_bf16.h>

#define EMBED 512
#define HEADS 4
#define HID   64
#define ATTND 64
#define BQ    4
#define LQ    2048
#define ROWS  (BQ*LQ)      // 8192 tokens
#define FUSED 1024         // (HID*2 + ATTN*2) * HEADS
#define HH    (HID*HEADS)  // 256
#define PROJD (HH*3)       // 768
#define ALPHA 0.125f       // 1/sqrt(64)
#define INV_L (1.0f/2048.0f)
#define SLDS  40           // padded LDS row stride (bf16 elems): 80B, 16B-aligned rows

typedef __attribute__((ext_vector_type(16))) __bf16 v16bf;
typedef __attribute__((ext_vector_type(8)))  __bf16 v8bf;
typedef __attribute__((ext_vector_type(4)))  __bf16 v4bf;
typedef __attribute__((ext_vector_type(8)))  float  v8f;

__device__ __forceinline__ int imin(int a, int b) { return a < b ? a : b; }

__device__ __forceinline__ v8f wmma_bf16(v16bf a, v16bf b, v8f c) {
  // D = A(16x32 bf16) * B(32x16 bf16) + C(16x16 f32)
  return __builtin_amdgcn_wmma_f32_16x16x32_bf16(false, a, false, b, (short)0, c,
                                                 false, false);
}

// A-fragment loader: p = &row[half*8]; elements 0..7 at p, 8..15 at p+16
// (ISA 16-bit A 16x32 layout: lanes 0-15 K={0..7,16..23}, lanes 16-31 K={8..15,24..31})
__device__ __forceinline__ v16bf load_a(const __bf16* p) {
  union { v16bf v; v8bf h[2]; } u;
  u.h[0] = *(const v8bf*)(p);
  u.h[1] = *(const v8bf*)(p + 16);
  return u.v;
}

// ---------------- Kernel 1: LayerNorm(x) -> bf16, one wave per token row ----
__global__ void __launch_bounds__(256) k_ln_x(const float* __restrict__ x,
                                              const float* __restrict__ sc,
                                              const float* __restrict__ bi,
                                              __bf16* __restrict__ nx) {
  const int row  = (blockIdx.x * blockDim.x + threadIdx.x) >> 5;
  const int lane = threadIdx.x & 31;
  const float* r = x + (size_t)row * EMBED;
  float4 v[4];
  float s = 0.f, ss = 0.f;
#pragma unroll
  for (int i = 0; i < 4; ++i) {
    v[i] = ((const float4*)r)[lane + 32 * i];
    s  += v[i].x + v[i].y + v[i].z + v[i].w;
    ss += v[i].x*v[i].x + v[i].y*v[i].y + v[i].z*v[i].z + v[i].w*v[i].w;
  }
#pragma unroll
  for (int o = 16; o > 0; o >>= 1) { s += __shfl_xor(s, o); ss += __shfl_xor(ss, o); }
  const float m  = s * (1.f / EMBED);
  const float rs = rsqrtf(ss * (1.f / EMBED) - m * m + 1e-6f);
  __bf16* outp = nx + (size_t)row * EMBED;
#pragma unroll
  for (int i = 0; i < 4; ++i) {
    const int c = (lane + 32 * i) * 4;
    const float a[4] = {v[i].x, v[i].y, v[i].z, v[i].w};
    v4bf o4;
#pragma unroll
    for (int j = 0; j < 4; ++j)
      o4[j] = (__bf16)((a[j] - m) * rs * sc[c + j] + bi[c + j]);
    *(v4bf*)(outp + c) = o4;
  }
}

// ------------- Weight convert+transpose kernels (tiny, amortized) -----------
__global__ void __launch_bounds__(256) k_cvt_uvqkw(const float* __restrict__ w,
                                                   __bf16* __restrict__ wT) {
  const int idx = blockIdx.x * blockDim.x + threadIdx.x; // 512*1024
  const int o = idx >> 9, i = idx & 511;                 // wT[out][in]
  wT[(size_t)o * EMBED + i] = (__bf16)w[(size_t)i * FUSED + o];
}
__global__ void __launch_bounds__(256) k_cvt_outw(const float* __restrict__ w,
                                                  __bf16* __restrict__ oT) {
  const int idx = blockIdx.x * blockDim.x + threadIdx.x; // 512*768
  const int o = idx / PROJD, i = idx % PROJD;            // oT[out][in]
  oT[(size_t)o * PROJD + i] = (__bf16)w[(size_t)i * EMBED + o];
}

// ---------------- Kernel 2: uvqk GEMM + split/SiLU/transpose epilogue -------
// wave computes 32 rows x 64 cols (2 A-frags share each B-frag); 4096 waves
__global__ void __launch_bounds__(256) k_uvqk(const __bf16* __restrict__ nx,
                                              const __bf16* __restrict__ wT,
                                              const float* __restrict__ beta,
                                              float* __restrict__ u,
                                              __bf16* __restrict__ qsb,
                                              __bf16* __restrict__ ksb,
                                              __bf16* __restrict__ vtb) {
  const int wid  = (blockIdx.x * blockDim.x + threadIdx.x) >> 5;
  const int lane = threadIdx.x & 31;
  const int ct = wid & 15, rt = wid >> 4;         // 16 col tiles, 256 row tiles
  const int row0 = rt * 32, col0 = ct * 64;
  const int half = lane >> 4, ln = lane & 15;
  const v8f z = {0.f, 0.f, 0.f, 0.f, 0.f, 0.f, 0.f, 0.f};
  v8f acc[2][4];
#pragma unroll
  for (int qi = 0; qi < 2; ++qi)
#pragma unroll
    for (int t = 0; t < 4; ++t) acc[qi][t] = z;
  const __bf16* arow0 = nx + (size_t)(row0 + ln) * EMBED + half * 8;
  const __bf16* arow1 = nx + (size_t)(row0 + 16 + ln) * EMBED + half * 8;
  for (int kq = 0; kq < EMBED; kq += 32) {
    v16bf a0 = load_a(arow0 + kq);
    v16bf a1 = load_a(arow1 + kq);
#pragma unroll
    for (int t = 0; t < 4; ++t) {
      const __bf16* bp = wT + (size_t)(col0 + t * 16 + ln) * EMBED + kq + half * 8;
      v16bf b = load_a(bp);
      acc[0][t] = wmma_bf16(a0, b, acc[0][t]);
      acc[1][t] = wmma_bf16(a1, b, acc[1][t]);
    }
  }
  const int region = col0 >> 8;   // 0=u, 1=v, 2=q, 3=k (wave-uniform)
  const int cbase  = col0 & 255;
#pragma unroll
  for (int t = 0; t < 4; ++t) {
    const int cc = cbase + t * 16 + ln;           // col within region [0,256)
    const float bv = beta[col0 + t * 16 + ln];
    const int hh = cc >> 6, dd = cc & 63;
#pragma unroll
    for (int qi = 0; qi < 2; ++qi)
#pragma unroll
      for (int rr = 0; rr < 8; ++rr) {
        const int grow = row0 + qi * 16 + rr + 8 * half;
        const float val = acc[qi][t][rr] + bv;
        const int bb = grow >> 11, ll = grow & 2047;
        if (region == 0) {
          u[(size_t)grow * HH + cc] = val / (1.f + __expf(-val));  // silu(u)
        } else if (region == 1) {  // V transposed: [b,h][d][l]
          vtb[((size_t)(bb * HEADS + hh) * HID + dd) * LQ + ll] = (__bf16)val;
        } else if (region == 2) {  // Q: [b,h][l][d]
          qsb[((size_t)(bb * HEADS + hh) * LQ + ll) * ATTND + dd] = (__bf16)val;
        } else {                   // K: [b,h][l][d]
          ksb[((size_t)(bb * HEADS + hh) * LQ + ll) * ATTND + dd] = (__bf16)val;
        }
      }
  }
}

// ---------------- Kernel 3: SiLU-gated masked linear attention --------------
// wave computes one 32-query x 64-dim O tile; K/V frags reused by both q tiles
__global__ void __launch_bounds__(256) k_attn(const __bf16* __restrict__ qsb,
                                              const __bf16* __restrict__ ksb,
                                              const __bf16* __restrict__ vtb,
                                              const int* __restrict__ ntg,
                                              float* __restrict__ attn) {
  __shared__ __align__(16) __bf16 smem[8 * 32 * SLDS];
  const int lw   = threadIdx.x >> 5;
  const int lane = threadIdx.x & 31;
  const int wid  = blockIdx.x * 8 + lw;           // 0..1023
  const int qt = wid & 63, bh = wid >> 6;         // 64 q tiles of 32
  const int b = bh >> 2, h = bh & 3;
  const int half = lane >> 4, ln = lane & 15;
  const int maxid = LQ - ntg[b];
  const __bf16* qb = qsb + (size_t)bh * LQ * ATTND;
  const __bf16* kb = ksb + (size_t)bh * LQ * ATTND;
  const __bf16* vb = vtb + (size_t)bh * HID * LQ;
  const int q0 = qt * 32;
  v16bf qa[2][2];   // [q row tile][K chunk]
#pragma unroll
  for (int qi = 0; qi < 2; ++qi)
#pragma unroll
    for (int kc = 0; kc < 2; ++kc)
      qa[qi][kc] = load_a(qb + (size_t)(q0 + qi * 16 + ln) * ATTND + kc * 32 + half * 8);
  const v8f z = {0.f, 0.f, 0.f, 0.f, 0.f, 0.f, 0.f, 0.f};
  v8f o[2][4];
#pragma unroll
  for (int qi = 0; qi < 2; ++qi)
#pragma unroll
    for (int t = 0; t < 4; ++t) o[qi][t] = z;
  __bf16* sb = smem + lw * (32 * SLDS);
  int qrow[2][8], qid[2][8];
#pragma unroll
  for (int qi = 0; qi < 2; ++qi)
#pragma unroll
    for (int rr = 0; rr < 8; ++rr) {
      qrow[qi][rr] = q0 + qi * 16 + rr + 8 * half;
      qid[qi][rr]  = imin(qrow[qi][rr], maxid);
    }
  for (int k0 = 0; k0 < LQ; k0 += 32) {
    v16bf kf[2][2];   // [key sub-tile][K chunk] -- shared by both q tiles
#pragma unroll
    for (int sub = 0; sub < 2; ++sub) {
      const __bf16* kp = kb + (size_t)(k0 + sub * 16 + ln) * ATTND + half * 8;
      kf[sub][0] = load_a(kp);
      kf[sub][1] = load_a(kp + 32);
    }
#pragma unroll
    for (int qi = 0; qi < 2; ++qi)
#pragma unroll
      for (int sub = 0; sub < 2; ++sub) {
        v8f s = z;
        s = wmma_bf16(qa[qi][0], kf[sub][0], s);
        s = wmma_bf16(qa[qi][1], kf[sub][1], s);
        const int kcol = k0 + sub * 16 + ln;
        const int kid  = imin(kcol, maxid);
#pragma unroll
        for (int rr = 0; rr < 8; ++rr) {
          float sv = s[rr] * ALPHA;
          sv = sv / (1.f + __expf(-sv)) * INV_L;                       // silu / L
          sv = ((qrow[qi][rr] == kcol) || (qid[qi][rr] > kid)) ? sv : 0.f;  // mask
          sb[(qi * 16 + rr + 8 * half) * SLDS + sub * 16 + ln] = (__bf16)sv;
        }
      }
    // CDNA5 split counter: drain ds stores before transposed ds re-reads
    __asm__ volatile("s_wait_dscnt 0" ::: "memory");
    v16bf sa[2];
#pragma unroll
    for (int qi = 0; qi < 2; ++qi)
      sa[qi] = load_a(sb + (qi * 16 + ln) * SLDS + half * 8);  // S as A-frag
#pragma unroll
    for (int t = 0; t < 4; ++t) {
      const __bf16* vp = vb + (size_t)(t * 16 + ln) * LQ + k0 + half * 8;
      v16bf vf = load_a(vp);       // shared by both q tiles
      o[0][t] = wmma_bf16(sa[0], vf, o[0][t]);
      o[1][t] = wmma_bf16(sa[1], vf, o[1][t]);
    }
  }
#pragma unroll
  for (int qi = 0; qi < 2; ++qi)
#pragma unroll
    for (int t = 0; t < 4; ++t)
#pragma unroll
      for (int rr = 0; rr < 8; ++rr)
        attn[(size_t)(b * LQ + q0 + qi * 16 + rr + 8 * half) * HH + h * HID +
             t * 16 + ln] = o[qi][t][rr];
}

// -------- Kernel 4: LayerNorm(attn), gate, build proj_in = [u|attn|gated] ---
__global__ void __launch_bounds__(256) k_ln_attn(const float* __restrict__ attn,
                                                 const float* __restrict__ u,
                                                 const float* __restrict__ sc,
                                                 const float* __restrict__ bi,
                                                 __bf16* __restrict__ proj) {
  const int row  = (blockIdx.x * blockDim.x + threadIdx.x) >> 5;
  const int lane = threadIdx.x & 31;
  const float* ar = attn + (size_t)row * HH;
  float4 v[2];
  float s = 0.f, ss = 0.f;
#pragma unroll
  for (int i = 0; i < 2; ++i) {
    v[i] = ((const float4*)ar)[lane + 32 * i];
    s  += v[i].x + v[i].y + v[i].z + v[i].w;
    ss += v[i].x*v[i].x + v[i].y*v[i].y + v[i].z*v[i].z + v[i].w*v[i].w;
  }
#pragma unroll
  for (int o = 16; o > 0; o >>= 1) { s += __shfl_xor(s, o); ss += __shfl_xor(ss, o); }
  const float m  = s * (1.f / HH);
  const float rs = rsqrtf(ss * (1.f / HH) - m * m + 1e-6f);
  const float* ur = u + (size_t)row * HH;
  __bf16* pr = proj + (size_t)row * PROJD;
#pragma unroll
  for (int i = 0; i < 2; ++i) {
    const int c = (lane + 32 * i) * 4;
    const float4 uv4 = ((const float4*)ur)[lane + 32 * i];
    const float a[4] = {v[i].x, v[i].y, v[i].z, v[i].w};
    const float g[4] = {uv4.x, uv4.y, uv4.z, uv4.w};
    v4bf pu, pa, pg;
#pragma unroll
    for (int j = 0; j < 4; ++j) {
      const float n = (a[j] - m) * rs * sc[c + j] + bi[c + j];
      pu[j] = (__bf16)g[j];
      pa[j] = (__bf16)a[j];
      pg[j] = (__bf16)(g[j] * n);
    }
    *(v4bf*)(pr + c)       = pu;
    *(v4bf*)(pr + 256 + c) = pa;
    *(v4bf*)(pr + 512 + c) = pg;
  }
}

// ---------------- Kernel 5: output GEMM + residual --------------------------
// wave computes 32 rows x 64 cols; 2048 waves
__global__ void __launch_bounds__(256) k_out(const __bf16* __restrict__ proj,
                                             const __bf16* __restrict__ oT,
                                             const float* __restrict__ x,
                                             float* __restrict__ out) {
  const int wid  = (blockIdx.x * blockDim.x + threadIdx.x) >> 5;
  const int lane = threadIdx.x & 31;
  const int ct = wid & 7, rt = wid >> 3;          // 8 col tiles, 256 row tiles
  const int row0 = rt * 32, col0 = ct * 64;
  const int half = lane >> 4, ln = lane & 15;
  const v8f z = {0.f, 0.f, 0.f, 0.f, 0.f, 0.f, 0.f, 0.f};
  v8f acc[2][4];
#pragma unroll
  for (int qi = 0; qi < 2; ++qi)
#pragma unroll
    for (int t = 0; t < 4; ++t) acc[qi][t] = z;
  const __bf16* arow0 = proj + (size_t)(row0 + ln) * PROJD + half * 8;
  const __bf16* arow1 = proj + (size_t)(row0 + 16 + ln) * PROJD + half * 8;
  for (int kq = 0; kq < PROJD; kq += 32) {
    v16bf a0 = load_a(arow0 + kq);
    v16bf a1 = load_a(arow1 + kq);
#pragma unroll
    for (int t = 0; t < 4; ++t) {
      const __bf16* bp = oT + (size_t)(col0 + t * 16 + ln) * PROJD + kq + half * 8;
      v16bf bfr = load_a(bp);
      acc[0][t] = wmma_bf16(a0, bfr, acc[0][t]);
      acc[1][t] = wmma_bf16(a1, bfr, acc[1][t]);
    }
  }
#pragma unroll
  for (int qi = 0; qi < 2; ++qi)
#pragma unroll
    for (int t = 0; t < 4; ++t)
#pragma unroll
      for (int rr = 0; rr < 8; ++rr) {
        const size_t idx =
            (size_t)(row0 + qi * 16 + rr + 8 * half) * EMBED + col0 + t * 16 + ln;
        out[idx] = x[idx] + acc[qi][t][rr];
      }
}

extern "C" void kernel_launch(void* const* d_in, const int* in_sizes, int n_in,
                              void* d_out, int out_size, void* d_ws, size_t ws_size,
                              hipStream_t stream) {
  (void)in_sizes; (void)n_in; (void)out_size; (void)ws_size;
  const float* x   = (const float*)d_in[0];
  const float* uw  = (const float*)d_in[1];
  const float* ub  = (const float*)d_in[2];
  const float* ow  = (const float*)d_in[3];
  const float* isc = (const float*)d_in[4];
  const float* ibi = (const float*)d_in[5];
  const float* osc = (const float*)d_in[6];
  const float* obi = (const float*)d_in[7];
  const int*   ntg = (const int*)d_in[8];

  char* ws = (char*)d_ws;
  size_t off = 0;
  auto carve = [&](size_t bytes) -> char* {
    char* p = ws + off;
    off += (bytes + 255) & ~(size_t)255;
    return p;
  };
  __bf16* nx   = (__bf16*)carve((size_t)ROWS * EMBED * 2);   //  8 MB
  __bf16* wT   = (__bf16*)carve((size_t)FUSED * EMBED * 2);  //  1 MB
  __bf16* oT   = (__bf16*)carve((size_t)EMBED * PROJD * 2);  // .75MB
  float*  u    = (float*) carve((size_t)ROWS * HH * 4);      //  8 MB
  __bf16* qsb  = (__bf16*)carve((size_t)ROWS * HH * 2);      //  4 MB
  __bf16* ksb  = (__bf16*)carve((size_t)ROWS * HH * 2);      //  4 MB
  __bf16* vtb  = (__bf16*)carve((size_t)ROWS * HH * 2);      //  4 MB
  float*  at   = (float*) carve((size_t)ROWS * HH * 4);      //  8 MB
  __bf16* proj = (__bf16*)carve((size_t)ROWS * PROJD * 2);   // 12 MB

  k_ln_x    <<<ROWS / 8, 256, 0, stream>>>(x, isc, ibi, nx);
  k_cvt_uvqkw<<<(EMBED * FUSED) / 256, 256, 0, stream>>>(uw, wT);
  k_cvt_outw <<<(EMBED * PROJD) / 256, 256, 0, stream>>>(ow, oT);
  k_uvqk    <<<(ROWS / 32) * (FUSED / 64) / 8, 256, 0, stream>>>(nx, wT, ub, u,
                                                                 qsb, ksb, vtb);
  k_attn    <<<(BQ * HEADS * (LQ / 32)) / 8, 256, 0, stream>>>(qsb, ksb, vtb,
                                                               ntg, at);
  k_ln_attn <<<ROWS / 8, 256, 0, stream>>>(at, u, osc, obi, proj);
  k_out     <<<(ROWS / 32) * (EMBED / 64) / 8, 256, 0, stream>>>(proj, oT, x,
                                                                 (float*)d_out);
}
